// MultiheadDiffAttn_29807073034391
// MI455X (gfx1250) — matrix operand order repaired and checked by reading, not compiled, not run on hardware
//
#include <hip/hip_runtime.h>
#include <hip/hip_bf16.h>
#include <stdint.h>
#include <math.h>

// ---------------- CDNA5 WMMA types ----------------
typedef __attribute__((ext_vector_type(16))) __bf16 v16bf;
typedef __attribute__((ext_vector_type(8)))  float  v8f;

#define EMBED 1024
#define TSEQ  2048
#define BATCH 2
#define NHEADS 8
#define HDIM  64
#define LOG2E 1.4426950408889634f

union FragBF { v16bf v; uint32_t u[8]; };

__device__ __forceinline__ uint16_t f2bf(float f) {
  uint32_t u = __builtin_bit_cast(uint32_t, f);
  u += 0x7fffu + ((u >> 16) & 1u);               // round-to-nearest-even
  return (uint16_t)(u >> 16);
}

// A-fragment K-offset (16x32 bf16): lanes 0-15 K={0,2,4,6}+.., lanes 16-31 +8; j>=4 adds 16
__device__ __forceinline__ int akoff(int j, int lane) {
  return ((j >> 2) << 4) + ((lane >> 4) << 3) + ((j & 3) << 1);
}
// B-fragment K-offset (32x16 bf16): same as A: lanes 0-15 K={0,2,4,6}+.., lanes 16-31 +8; j>=4 adds 16
__device__ __forceinline__ int bkoff(int j, int lane) {
  return ((j >> 2) << 4) + ((lane >> 4) << 3) + ((j & 3) << 1);
}

__device__ __forceinline__ v8f wmma_bf16(v16bf a, v16bf b, v8f c) {
  return __builtin_amdgcn_wmma_f32_16x16x32_bf16(false, a, false, b, (short)0, c, false, false);
}

// async global->LDS copy of 16 bytes (ASYNCcnt-tracked, no VGPR round trip)
__device__ __forceinline__ void async_copy_b128(const void* gptr, void* lptr) {
  uint32_t lds = (uint32_t)(uintptr_t)lptr;          // flat LDS addr: low 32 bits = LDS offset
  uint64_t ga  = (uint64_t)(uintptr_t)gptr;
  asm volatile("global_load_async_to_lds_b128 %0, %1, off"
               :: "v"(lds), "v"(ga) : "memory");
}
__device__ __forceinline__ void wait_asynccnt0() {
  asm volatile("s_wait_asynccnt 0" ::: "memory");
}

// ---------------- lambda_full reduction ----------------
__global__ void lambda_kernel(const float* __restrict__ lq1, const float* __restrict__ lk1,
                              const float* __restrict__ lq2, const float* __restrict__ lk2,
                              float* __restrict__ out) {
  __shared__ float s1[64], s2[64];
  int t = threadIdx.x;
  s1[t] = lq1[t] * lk1[t];
  s2[t] = lq2[t] * lk2[t];
  __syncthreads();
  if (t == 0) {
    float a = 0.f, b = 0.f;
    for (int i = 0; i < 64; ++i) { a += s1[i]; b += s2[i]; }
    *out = expf(a) - expf(b) + 0.8f;
  }
}

// ---------------- bf16 WMMA GEMM: C[M,N] = A[M,K] * B[N,K]^T ----------------
// M=4096 (grid.x=32), N=K=1024. A is fp32 or bf16; B always fp32 (weights); out bf16 or fp32.
template<bool ABF16, bool OUTF32>
__global__ __launch_bounds__(256) void gemm_nt_kernel(const void* __restrict__ Aptr,
                                                      const float* __restrict__ Bw,
                                                      void* __restrict__ Cptr, float scale) {
  __shared__ uint16_t As[128][72];
  __shared__ uint16_t Bs[128][72];
  const int tid = threadIdx.x;
  const int lane = tid & 31, wave = tid >> 5;
  const int bm = blockIdx.x * 128, bn = blockIdx.y * 128;
  const int wm = (wave & 3) * 32, wn = (wave >> 2) * 64;
  v8f acc[2][4] = {};
  const int ar = tid >> 1, ac0 = (tid & 1) * 32;

  for (int k0 = 0; k0 < EMBED; k0 += 64) {
    if (ABF16) {
      const uint16_t* Ab = (const uint16_t*)Aptr;
      #pragma unroll
      for (int i = 0; i < 32; i += 8)
        async_copy_b128(&Ab[(size_t)(bm + ar) * EMBED + k0 + ac0 + i], &As[ar][ac0 + i]);
    } else {
      const float* Af = (const float*)Aptr;
      #pragma unroll
      for (int i = 0; i < 32; ++i)
        As[ar][ac0 + i] = f2bf(Af[(size_t)(bm + ar) * EMBED + k0 + ac0 + i]);
    }
    #pragma unroll
    for (int i = 0; i < 32; ++i)
      Bs[ar][ac0 + i] = f2bf(Bw[(size_t)(bn + ar) * EMBED + k0 + ac0 + i]);
    if (ABF16) wait_asynccnt0();
    __syncthreads();

    #pragma unroll
    for (int ks = 0; ks < 64; ks += 32) {
      FragBF a[2], b[4];
      #pragma unroll
      for (int mi = 0; mi < 2; ++mi)
        #pragma unroll
        for (int j = 0; j < 8; ++j)
          a[mi].u[j] = *(const uint32_t*)&As[wm + mi * 16 + (lane & 15)][ks + akoff(j, lane)];
      #pragma unroll
      for (int ni = 0; ni < 4; ++ni)
        #pragma unroll
        for (int j = 0; j < 8; ++j)
          b[ni].u[j] = *(const uint32_t*)&Bs[wn + ni * 16 + (lane & 15)][ks + bkoff(j, lane)];
      #pragma unroll
      for (int mi = 0; mi < 2; ++mi)
        #pragma unroll
        for (int ni = 0; ni < 4; ++ni)
          acc[mi][ni] = wmma_bf16(a[mi].v, b[ni].v, acc[mi][ni]);
    }
    __syncthreads();
  }

  #pragma unroll
  for (int mi = 0; mi < 2; ++mi)
    #pragma unroll
    for (int ni = 0; ni < 4; ++ni)
      #pragma unroll
      for (int j = 0; j < 8; ++j) {
        int row = bm + wm + mi * 16 + ((lane >> 4) << 3) + j;
        int col = bn + wn + ni * 16 + (lane & 15);
        float v = acc[mi][ni][j] * scale;
        if (OUTF32) ((float*)Cptr)[(size_t)row * EMBED + col] = v;
        else ((uint16_t*)Cptr)[(size_t)row * EMBED + col] = f2bf(v);
      }
}

// ---------------- fused flash differential attention ----------------
// grid: (T/16/8=16, B*H=16); block 256 = 8 waves; each wave owns a 16-row query tile.
__global__ __launch_bounds__(256) void diffattn_kernel(
    const uint16_t* __restrict__ Qh, const uint16_t* __restrict__ Kh,
    const uint16_t* __restrict__ Vh, const float* __restrict__ lamp,
    uint16_t* __restrict__ AO) {
  __shared__ uint16_t Kt[64][136];     // [s][d(0..127)] both half-heads
  __shared__ uint16_t VtT[128][72];    // transposed V tile: [d][s]
  __shared__ uint16_t Pb[8][16][72];   // per-wave P transpose staging
  const int tid = threadIdx.x, lane = tid & 31, wave = tid >> 5;
  const int b = blockIdx.y >> 3, h = blockIdx.y & 7;
  const int qrow0 = (blockIdx.x * 8 + wave) * 16;
  const size_t rowbase = (size_t)b * TSEQ;
  const int kcol0 = (2 * h) * HDIM;    // 128 contiguous cols (half-heads 2h,2h+1)
  const int vcol0 = h * 2 * HDIM;      // 128 contiguous cols (head h)
  const float lamf = lamp[0];

  // resident Q A-fragments: [half][k-frag]
  FragBF qa[2][2];
  #pragma unroll
  for (int half = 0; half < 2; ++half)
    #pragma unroll
    for (int kf = 0; kf < 2; ++kf)
      #pragma unroll
      for (int j = 0; j < 8; ++j)
        qa[half][kf].u[j] = *(const uint32_t*)&Qh[(rowbase + qrow0 + (lane & 15)) * EMBED +
                                                  kcol0 + half * 64 + kf * 32 + akoff(j, lane)];

  v8f o1[8] = {}, o2[8] = {};
  float m1[8], l1[8], m2[8], l2[8];
  #pragma unroll
  for (int j = 0; j < 8; ++j) { m1[j] = -INFINITY; m2[j] = -INFINITY; l1[j] = 0.f; l2[j] = 0.f; }

  const int r = tid >> 2, cpart = (tid & 3) * 32;
  for (int s0 = 0; s0 < TSEQ; s0 += 64) {
    // stage K tile via ASYNCcnt-tracked global->LDS DMA (pure copy, no transform)
    #pragma unroll
    for (int c = 0; c < 32; c += 8)
      async_copy_b128(&Kh[(rowbase + s0 + r) * EMBED + kcol0 + cpart + c],
                      &Kt[r][cpart + c]);
    // stage V tile transposed (transform -> register path)
    #pragma unroll
    for (int c = 0; c < 32; c += 8) {
      uint4 vv = *(const uint4*)&Vh[(rowbase + s0 + r) * EMBED + vcol0 + cpart + c];
      uint32_t w0 = vv.x, w1 = vv.y, w2 = vv.z, w3 = vv.w;
      VtT[cpart + c + 0][r] = (uint16_t)w0; VtT[cpart + c + 1][r] = (uint16_t)(w0 >> 16);
      VtT[cpart + c + 2][r] = (uint16_t)w1; VtT[cpart + c + 3][r] = (uint16_t)(w1 >> 16);
      VtT[cpart + c + 4][r] = (uint16_t)w2; VtT[cpart + c + 5][r] = (uint16_t)(w2 >> 16);
      VtT[cpart + c + 6][r] = (uint16_t)w3; VtT[cpart + c + 7][r] = (uint16_t)(w3 >> 16);
    }
    if (s0 + 64 < TSEQ) {   // global_prefetch_b8 next tile
      __builtin_prefetch(&Kh[(rowbase + s0 + 64 + r) * EMBED + kcol0 + cpart], 0, 1);
      __builtin_prefetch(&Vh[(rowbase + s0 + 64 + r) * EMBED + vcol0 + cpart], 0, 1);
    }
    wait_asynccnt0();
    __syncthreads();

    #pragma unroll
    for (int half = 0; half < 2; ++half) {
      v8f* o = half ? o2 : o1;
      float* mm = half ? m2 : m1;
      float* ll = half ? l2 : l1;

      // scores: Q(16x64) . K^T -> 16 x 64 in four 16x16 D-fragments
      v8f sc[4];
      #pragma unroll
      for (int ns = 0; ns < 4; ++ns) {
        v8f c = {};
        #pragma unroll
        for (int kf = 0; kf < 2; ++kf) {
          FragBF bf;
          #pragma unroll
          for (int j = 0; j < 8; ++j)
            bf.u[j] = *(const uint32_t*)&Kt[ns * 16 + (lane & 15)]
                                           [half * 64 + kf * 32 + bkoff(j, lane)];
          c = wmma_bf16(qa[half][kf].v, bf.v, c);
        }
        sc[ns] = c;
      }

      // online softmax: rows live at (reg j, lane-half); reduce across 16-lane group
      float fac[8], ps[8];
      #pragma unroll
      for (int j = 0; j < 8; ++j) {
        float t = fmaxf(fmaxf(sc[0][j], sc[1][j]), fmaxf(sc[2][j], sc[3][j]));
        t = fmaxf(t, __shfl_xor(t, 1, 32));
        t = fmaxf(t, __shfl_xor(t, 2, 32));
        t = fmaxf(t, __shfl_xor(t, 4, 32));
        t = fmaxf(t, __shfl_xor(t, 8, 32));
        float mo = mm[j];
        float mn = fmaxf(mo, t);
        fac[j] = exp2f((mo - mn) * LOG2E);
        mm[j] = mn;
        ps[j] = 0.f;
      }
      #pragma unroll
      for (int ns = 0; ns < 4; ++ns)
        #pragma unroll
        for (int j = 0; j < 8; ++j) {
          float p = exp2f((sc[ns][j] - mm[j]) * LOG2E);
          sc[ns][j] = p;
          ps[j] += p;
        }
      #pragma unroll
      for (int j = 0; j < 8; ++j) {
        float t = ps[j];
        t += __shfl_xor(t, 1, 32);
        t += __shfl_xor(t, 2, 32);
        t += __shfl_xor(t, 4, 32);
        t += __shfl_xor(t, 8, 32);
        ll[j] = ll[j] * fac[j] + t;
      }
      #pragma unroll
      for (int ni = 0; ni < 8; ++ni)
        #pragma unroll
        for (int j = 0; j < 8; ++j)
          o[ni][j] *= fac[j];

      // D-layout -> A-fragment layout via per-wave LDS round trip
      #pragma unroll
      for (int ns = 0; ns < 4; ++ns)
        #pragma unroll
        for (int j = 0; j < 8; ++j)
          Pb[wave][((lane >> 4) << 3) + j][ns * 16 + (lane & 15)] = f2bf(sc[ns][j]);
      asm volatile("s_wait_dscnt 0" ::: "memory");
      FragBF pa[2];
      #pragma unroll
      for (int kf = 0; kf < 2; ++kf)
        #pragma unroll
        for (int j = 0; j < 8; ++j)
          pa[kf].u[j] = *(const uint32_t*)&Pb[wave][lane & 15][kf * 32 + akoff(j, lane)];

      // O += P(16x64) . V(64x128)
      #pragma unroll
      for (int ni = 0; ni < 8; ++ni) {
        #pragma unroll
        for (int kf = 0; kf < 2; ++kf) {
          FragBF vb;
          #pragma unroll
          for (int j = 0; j < 8; ++j)
            vb.u[j] = *(const uint32_t*)&VtT[ni * 16 + (lane & 15)][kf * 32 + bkoff(j, lane)];
          o[ni] = wmma_bf16(pa[kf].v, vb.v, o[ni]);
        }
      }
    }
    __syncthreads();
  }

  // combine halves, RMSNorm over 128 dims, scale by (1 - lambda_init)
  float ss[8];
  #pragma unroll
  for (int j = 0; j < 8; ++j) ss[j] = 0.f;
  #pragma unroll
  for (int ni = 0; ni < 8; ++ni)
    #pragma unroll
    for (int j = 0; j < 8; ++j) {
      float cmb = o1[ni][j] / l1[j] - lamf * (o2[ni][j] / l2[j]);
      o1[ni][j] = cmb;
      ss[j] += cmb * cmb;
    }
  #pragma unroll
  for (int j = 0; j < 8; ++j) {
    float t = ss[j];
    t += __shfl_xor(t, 1, 32);
    t += __shfl_xor(t, 2, 32);
    t += __shfl_xor(t, 4, 32);
    t += __shfl_xor(t, 8, 32);
    ss[j] = rsqrtf(t * (1.0f / 128.0f) + 1e-5f) * 0.2f;
  }
  #pragma unroll
  for (int ni = 0; ni < 8; ++ni)
    #pragma unroll
    for (int j = 0; j < 8; ++j) {
      size_t row = rowbase + qrow0 + ((lane >> 4) << 3) + j;
      int col = vcol0 + ni * 16 + (lane & 15);
      AO[row * EMBED + col] = f2bf(o1[ni][j] * ss[j]);
    }
}

// ---------------- host-side launcher ----------------
extern "C" void kernel_launch(void* const* d_in, const int* in_sizes, int n_in,
                              void* d_out, int out_size, void* d_ws, size_t ws_size,
                              hipStream_t stream) {
  (void)in_sizes; (void)n_in; (void)out_size; (void)ws_size;
  const float* q   = (const float*)d_in[0];
  const float* k   = (const float*)d_in[1];
  const float* v   = (const float*)d_in[2];
  const float* Wq  = (const float*)d_in[3];
  const float* Wk  = (const float*)d_in[4];
  const float* Wv  = (const float*)d_in[5];
  const float* Wo  = (const float*)d_in[6];
  const float* lq1 = (const float*)d_in[7];
  const float* lk1 = (const float*)d_in[8];
  const float* lq2 = (const float*)d_in[9];
  const float* lk2 = (const float*)d_in[10];

  char* ws = (char*)d_ws;
  const size_t SZ = (size_t)BATCH * TSEQ * EMBED * sizeof(uint16_t);  // 8 MiB
  uint16_t* Qh = (uint16_t*)(ws);
  uint16_t* Kh = (uint16_t*)(ws + SZ);
  uint16_t* Vh = (uint16_t*)(ws + 2 * SZ);
  uint16_t* AO = (uint16_t*)(ws + 3 * SZ);
  float*    lam = (float*)(ws + 4 * SZ);

  lambda_kernel<<<1, 64, 0, stream>>>(lq1, lk1, lq2, lk2, lam);

  dim3 gg(32, 8);  // M=4096 / 128, N=1024 / 128
  gemm_nt_kernel<false, false><<<gg, 256, 0, stream>>>(q, Wq, Qh, 0.125f); // 1/sqrt(64)
  gemm_nt_kernel<false, false><<<gg, 256, 0, stream>>>(k, Wk, Kh, 1.0f);
  gemm_nt_kernel<false, false><<<gg, 256, 0, stream>>>(v, Wv, Vh, 1.0f);

  diffattn_kernel<<<dim3(16, 16), 256, 0, stream>>>(Qh, Kh, Vh, lam, AO);

  gemm_nt_kernel<true, true><<<gg, 256, 0, stream>>>(AO, Wo, d_out, 1.0f);
}
